// HyperedgeConstruction_14826227105919
// MI455X (gfx1250) — matrix operand order repaired and checked by reading, not compiled, not run on hardware
//
#include <hip/hip_runtime.h>
#include <hip/hip_bf16.h>
#include <math.h>

// Problem constants (from reference): B=8, C=768, H=W=64 -> N=4096, K=256
#define BB 8
#define CC 768
#define NN 4096
#define KK 256
#define EPSV 1e-12f

typedef __bf16 v16bf __attribute__((ext_vector_type(16)));
typedef __bf16 v8bf  __attribute__((ext_vector_type(8)));
typedef float  v8f   __attribute__((ext_vector_type(8)));
typedef int    v4i   __attribute__((ext_vector_type(4)));

static __device__ __forceinline__ unsigned short f32_to_bf16(float f) {
    unsigned int u = __float_as_uint(f);
    u += 0x7FFFu + ((u >> 16) & 1u);   // round-to-nearest-even
    return (unsigned short)(u >> 16);
}

// --- CDNA5 async global->LDS copy (ASYNCcnt) with portable fallbacks --------
#if __has_builtin(__builtin_amdgcn_global_load_async_to_lds_b128)
static __device__ __forceinline__ void async_copy_b128(const unsigned short* g,
                                                       unsigned short* l) {
    __builtin_amdgcn_global_load_async_to_lds_b128(
        (__attribute__((address_space(1))) v4i*)g,
        (__attribute__((address_space(3))) v4i*)l, 0, 0);
}
#else
static __device__ __forceinline__ void async_copy_b128(const unsigned short* g,
                                                       unsigned short* l) {
    unsigned lds_off =
        (unsigned)(uintptr_t)(__attribute__((address_space(3))) unsigned short*)l;
    asm volatile("global_load_async_to_lds_b128 %0, %1, off"
                 :: "v"(lds_off), "v"(g) : "memory");
}
#endif

#if __has_builtin(__builtin_amdgcn_s_wait_asynccnt)
#define WAIT_ASYNC() __builtin_amdgcn_s_wait_asynccnt(0)
#else
#define WAIT_ASYNC() asm volatile("s_wait_asynccnt 0x0" ::: "memory")
#endif

// ---------------------------------------------------------------------------
// K1a: centroid inverse norms.  centroids layout (B,C,K); reduce over C.
__global__ void centroid_norms_kernel(const float* __restrict__ centroids,
                                      float* __restrict__ inv_cn) {
    int e = blockIdx.x * blockDim.x + threadIdx.x;   // e in [0, B*K)
    int b = e / KK, k = e % KK;
    const float* p = centroids + (size_t)b * CC * KK + k;
    float s = 0.f;
    #pragma unroll 8
    for (int c = 0; c < CC; ++c) s += p[(size_t)c * KK] * p[(size_t)c * KK];
    inv_cn[e] = 1.f / fmaxf(sqrtf(s), EPSV);
}

// K1b: normalized centroids -> bf16, transposed to (B,K,C) for WMMA A-operand.
__global__ void centroid_cast_kernel(const float* __restrict__ centroids,
                                     const float* __restrict__ inv_cn,
                                     unsigned short* __restrict__ cn) {
    int e = blockIdx.x * blockDim.x + threadIdx.x;   // e in [0, B*C*K)
    int b = e / (CC * KK);
    int r = e % (CC * KK);
    int c = r / KK, k = r % KK;
    float v = centroids[e] * inv_cn[b * KK + k];     // coalesced read
    cn[((size_t)b * KK + k) * CC + c] = f32_to_bf16(v);
}

// K2a: token inverse norms. x layout (B,C,N); reduce over C, thread per (b,n).
__global__ void token_norms_kernel(const float* __restrict__ x,
                                   float* __restrict__ inv_tn) {
    int e = blockIdx.x * blockDim.x + threadIdx.x;   // e in [0, B*N)
    int b = e / NN, n = e % NN;
    const float* p = x + (size_t)b * CC * NN + n;
    float s = 0.f;
    #pragma unroll 8
    for (int c = 0; c < CC; ++c) s += p[(size_t)c * NN] * p[(size_t)c * NN];
    inv_tn[e] = 1.f / fmaxf(sqrtf(s), EPSV);
}

// K2b: 64x64 LDS tile transpose: x (B,C,N) f32 -> xn (B,N,C) bf16 (normalized).
__global__ void token_transpose_kernel(const float* __restrict__ x,
                                       const float* __restrict__ inv_tn,
                                       unsigned short* __restrict__ xn) {
    __shared__ float tile[64][65];
    int b  = blockIdx.z;
    int c0 = blockIdx.y * 64;
    int n0 = blockIdx.x * 64;
    int tx = threadIdx.x & 63, ty = threadIdx.x >> 6;   // 256 threads
    #pragma unroll
    for (int i = 0; i < 16; ++i) {
        int cl = ty + 4 * i;
        tile[cl][tx] = x[((size_t)b * CC + c0 + cl) * NN + n0 + tx];
    }
    __syncthreads();
    #pragma unroll
    for (int i = 0; i < 16; ++i) {
        int nl = ty + 4 * i;
        float v = tile[tx][nl] * inv_tn[b * NN + n0 + nl];
        xn[((size_t)b * NN + n0 + nl) * CC + c0 + tx] = f32_to_bf16(v);
    }
}

// ---------------------------------------------------------------------------
// K3: cos-similarity GEMM + sigmoid.
// 256-thread block = 8 waves; block computes a 128(M) x 64(N) tile of
// sim = sigmoid(beta + alpha * cn(K,C) x xn(N,C)^T).
// The shared B-tile (64 tokens x 32-K chunk, 4 KB bf16) is staged in
// double-buffered LDS via async global->LDS copies (ASYNCcnt), overlapped with
// the current step's v_wmma_f32_16x16x32_bf16 ops. Padded 80-byte LDS row
// stride keeps the ds_load_b128 fragment reads bank-conflict free.
#define BROW 40   // LDS row stride in halfs (64 data + 16 pad bytes)
#define BTILE (64 * BROW)

__global__ void sim_gemm_kernel(const unsigned short* __restrict__ cn,
                                const unsigned short* __restrict__ xn,
                                const float* __restrict__ sim_alpha,
                                const float* __restrict__ sim_beta,
                                float* __restrict__ simout) {
    __shared__ __align__(16) unsigned short bsm[2 * BTILE];

    const int t    = threadIdx.x;          // 0..255
    const int lane = t & 31;               // EXEC all ones within wave (WMMA req.)
    const int wav  = t >> 5;               // 0..7
    const int half = lane >> 4;
    const int l16  = lane & 15;
    const int b  = blockIdx.z;
    const int m0 = blockIdx.y * 128 + wav * 16;   // this wave's centroid tile
    const int n0 = blockIdx.x * 64;               // block's token tile

    // Cooperative B-tile stage: 256 threads x 16B = 4 KB per K-step.
    const int srow = t >> 2;               // 0..63  (token row within tile)
    const int sseg = t & 3;                // 0..3   (16-byte segment)
    const unsigned short* gbase =
        xn + ((size_t)b * NN + n0 + srow) * CC + sseg * 8;
    unsigned short* lbase = bsm + srow * BROW + sseg * 8;

    // A: 16x32 bf16 fragment (ISA 7.12.2): lane row M=l16,
    // a[0..7]=K(kk+half*8+i), a[8..15]=K(kk+16+half*8+i).
    const unsigned short* Abase = cn + ((size_t)b * KK + (m0 + l16)) * CC;

    v8f acc[4];
    #pragma unroll
    for (int nt = 0; nt < 4; ++nt) acc[nt] = (v8f){0.f,0.f,0.f,0.f,0.f,0.f,0.f,0.f};

    // Prologue: stage first B-tile (kk = 0) into buffer 0.
    async_copy_b128(gbase, lbase);
    WAIT_ASYNC();
    __syncthreads();

    int cbuf = 0;
    for (int kk = 0; kk < CC; kk += 32) {
        // Kick off next tile into the other buffer; overlaps with WMMAs below.
        if (kk + 32 < CC)
            async_copy_b128(gbase + (kk + 32), bsm + (cbuf ^ 1) * BTILE +
                            (lbase - bsm));
        // A fragment from global (per-wave row, L2 resident).
        v8bf alo = *reinterpret_cast<const v8bf*>(Abase + kk + half * 8);
        v8bf ahi = *reinterpret_cast<const v8bf*>(Abase + kk + 16 + half * 8);
        v16bf afrag = __builtin_shufflevector(alo, ahi,
                0,1,2,3,4,5,6,7,8,9,10,11,12,13,14,15);
        const unsigned short* bt = bsm + cbuf * BTILE;
        #pragma unroll
        for (int nt = 0; nt < 4; ++nt) {
            // b[0..7] = xn[token][kk + half*8 + i], b[8..15] = xn[token][kk + 16 + half*8 + i]
            const unsigned short* Bp = bt + (nt * 16 + l16) * BROW + half * 8;
            v8bf b0 = *reinterpret_cast<const v8bf*>(Bp);
            v8bf b1 = *reinterpret_cast<const v8bf*>(Bp + 16);
            v16bf bfrag = __builtin_shufflevector(b0, b1,
                    0,1,2,3,4,5,6,7,8,9,10,11,12,13,14,15);
            acc[nt] = __builtin_amdgcn_wmma_f32_16x16x32_bf16(
                false, afrag, false, bfrag, (short)0, acc[nt], false, false);
        }
        WAIT_ASYNC();        // next tile resident
        __syncthreads();     // and everyone is done reading cbuf
        cbuf ^= 1;
    }

    const float alpha = sim_alpha[0];
    const float beta  = sim_beta[0];
    // D layout: d[r] = D[half*8 + r][l16]
    #pragma unroll
    for (int nt = 0; nt < 4; ++nt) {
        int col = n0 + nt * 16 + l16;
        float* op = simout + (size_t)b * KK * NN + col;
        #pragma unroll
        for (int r = 0; r < 8; ++r) {
            int row = m0 + half * 8 + r;
            float z = beta + alpha * acc[nt][r];
            op[(size_t)row * NN] = 1.f / (1.f + expf(-z));
        }
    }
}

// ---------------------------------------------------------------------------
// K4: per-token argmax over K=256 centroids; mask sim in place (first-occurrence
// ties via strict >, matching jnp.argmax); record assignment + weight.
__global__ void argmax_mask_kernel(float* __restrict__ simout,
                                   int* __restrict__ assign,
                                   float* __restrict__ weight) {
    int e = blockIdx.x * blockDim.x + threadIdx.x;   // e in [0, B*N)
    int b = e / NN, n = e % NN;
    float* col = simout + (size_t)b * KK * NN + n;
    float best = -INFINITY; int bi = 0;
    #pragma unroll 4
    for (int k = 0; k < KK; ++k) {
        float v = col[(size_t)k * NN];
        if (v > best) { best = v; bi = k; }
    }
    #pragma unroll 4
    for (int k = 0; k < KK; ++k)
        col[(size_t)k * NN] = (k == bi) ? best : 0.f;
    assign[e] = bi;
    weight[e] = best;
}

// ---------------------------------------------------------------------------
// K5: deterministic sparse aggregation + epilogue.
// Block per (k,b); assignments/weights staged in LDS; ascending-n register
// accumulation (no float atomics -> bitwise deterministic).
__global__ void aggregate_kernel(const float* __restrict__ x,
                                 const float* __restrict__ centroids,
                                 const int* __restrict__ assign,
                                 const float* __restrict__ weight,
                                 float* __restrict__ out_hyper) {
    __shared__ int   sa[NN];
    __shared__ float sw[NN];
    const int k = blockIdx.x, b = blockIdx.y, t = threadIdx.x;  // 256 threads
    for (int i = t; i < NN; i += 256) {
        sa[i] = assign[b * NN + i];
        sw[i] = weight[b * NN + i];
    }
    __syncthreads();

    const float* xb = x + (size_t)b * CC * NN;
    const int c0 = t, c1 = t + 256, c2 = t + 512;
    float a0 = 0.f, a1 = 0.f, a2 = 0.f;
    int cnt = 0;
    for (int n = 0; n < NN; ++n) {
        if (sa[n] == k) {                 // uniform across block -> cheap branch
            float wv = sw[n];
            ++cnt;
            a0 += wv * xb[(size_t)c0 * NN + n];
            a1 += wv * xb[(size_t)c1 * NN + n];
            a2 += wv * xb[(size_t)c2 * NN + n];
        }
    }
    const float inv = 1.f / (float)(cnt + 1);
    const float* cb = centroids + (size_t)b * CC * KK + k;   // ct[b,k,c] strided
    float* op = out_hyper + ((size_t)b * KK + k) * CC;
    op[c0] = (a0 + cb[(size_t)c0 * KK]) * inv;
    op[c1] = (a1 + cb[(size_t)c1 * KK]) * inv;
    op[c2] = (a2 + cb[(size_t)c2 * KK]) * inv;
}

// ---------------------------------------------------------------------------
extern "C" void kernel_launch(void* const* d_in, const int* in_sizes, int n_in,
                              void* d_out, int out_size, void* d_ws, size_t ws_size,
                              hipStream_t stream) {
    const float* x         = (const float*)d_in[0];   // (B,C,H,W) = (B,C,N)
    const float* centroids = (const float*)d_in[1];   // (B,C,K)
    const float* sim_alpha = (const float*)d_in[2];   // (1,)
    const float* sim_beta  = (const float*)d_in[3];   // (1,)

    float* out_hyper = (float*)d_out;                          // (B,K,C)
    float* simout    = (float*)d_out + (size_t)BB * KK * CC;   // (B,K,N)

    // Workspace carve-up (~54 MB)
    char* w = (char*)d_ws;
    size_t off = 0;
    unsigned short* cn = (unsigned short*)(w + off); off += (size_t)BB * KK * CC * 2;
    unsigned short* xn = (unsigned short*)(w + off); off += (size_t)BB * NN * CC * 2;
    float* inv_cn = (float*)(w + off); off += (size_t)BB * KK * 4;
    float* inv_tn = (float*)(w + off); off += (size_t)BB * NN * 4;
    int*   assign = (int*)  (w + off); off += (size_t)BB * NN * 4;
    float* weight = (float*)(w + off); off += (size_t)BB * NN * 4;
    (void)off; (void)ws_size; (void)in_sizes; (void)n_in; (void)out_size;

    centroid_norms_kernel<<<(BB * KK) / 256, 256, 0, stream>>>(centroids, inv_cn);
    centroid_cast_kernel<<<(BB * CC * KK) / 256, 256, 0, stream>>>(centroids, inv_cn, cn);
    token_norms_kernel<<<(BB * NN) / 256, 256, 0, stream>>>(x, inv_tn);
    token_transpose_kernel<<<dim3(NN / 64, CC / 64, BB), 256, 0, stream>>>(x, inv_tn, xn);
    sim_gemm_kernel<<<dim3(NN / 64, KK / 128, BB), 256, 0, stream>>>(
        cn, xn, sim_alpha, sim_beta, simout);
    argmax_mask_kernel<<<(BB * NN) / 256, 256, 0, stream>>>(simout, assign, weight);
    aggregate_kernel<<<dim3(KK, BB), 256, 0, stream>>>(x, centroids, assign, weight, out_hyper);
}
